// Decoder_2680059592732
// MI455X (gfx1250) — matrix operand order repaired and checked by reading, not compiled, not run on hardware
//
#include <hip/hip_runtime.h>
#include <hip/hip_bf16.h>

// ---------------------------------------------------------------------------
// Tacotron2-style decoder for MI455X (gfx1250).
// Heavy GEMMs use v_wmma_f32_16x16x32_bf16.  Both operands are staged in
// workspace in WMMA-fragment order ([ktile][tile][lane][16 bf16]) so the GEMM
// inner loop is pure b128 loads + WMMA.  200 sequential steps are issued as a
// host loop of kernels on `stream` (graph-capturable).
// ---------------------------------------------------------------------------

typedef __bf16 bf16_t;
typedef __attribute__((ext_vector_type(16))) __bf16 v16bf;
typedef __attribute__((ext_vector_type(8)))  float  v8f;

#define B_    32
#define TIN   256
#define TOUT  200
#define HID   256
#define EMB   256
#define ATT   128
#define ARNN  1024
#define DRNN  1024
#define PRE   128
#define NF    32
#define KS    31

// ---------------------------------------------------------------------------
// WMMA GEMM on pre-packed fragments:
//   C[M,N] (f32) = A[M,K] @ B[K,N] + bias[N]
// Af: A fragments, layout ((kt*(M/16)+mt)*32+lane)*16 bf16
// Bf: B fragments, layout ((kt*(N/16)+nt)*32+lane)*16 bf16
// One wave computes one 16x16 C tile; block = 128 threads = 4 waves.
// K must be a multiple of 32; bias must be a valid pointer.
// ---------------------------------------------------------------------------
__global__ __launch_bounds__(128) void wmma_gemm(
    const bf16_t* __restrict__ Af, const bf16_t* __restrict__ Bf,
    const float* __restrict__ bias, float* __restrict__ C,
    int M, int N, int K)
{
  const int lane = threadIdx.x & 31;
  const int wave = threadIdx.x >> 5;
  const int nt = blockIdx.x * 4 + wave;
  const int mt = blockIdx.y;
  if (nt * 16 >= N || mt * 16 >= M) return;   // wave-uniform: EXEC all-1s

  const int ktiles = K >> 5;
  const size_t astride = (size_t)(M >> 4) << 9;   // (M/16)*512 elements per kt
  const size_t bstride = (size_t)(N >> 4) << 9;
  const bf16_t* ap = Af + (((size_t)mt * 32) + lane) * 16;
  const bf16_t* bp = Bf + (((size_t)nt * 32) + lane) * 16;

  v8f acc = {};
#pragma unroll 4
  for (int kt = 0; kt < ktiles; ++kt) {
    v16bf a = *(const v16bf*)ap;
    v16bf b = *(const v16bf*)bp;
    acc = __builtin_amdgcn_wmma_f32_16x16x32_bf16(
        false, a, false, b, (short)0, acc, false, false);
    ap += astride;
    bp += bstride;
  }

  // C layout: VGPR r -> row mt*16 + r (+8 for upper lane-half), col = nt*16+lane%16
  const int col = nt * 16 + (lane & 15);
  const int r0  = mt * 16 + ((lane >> 4) << 3);
  const float bv = bias[col];
#pragma unroll
  for (int r = 0; r < 8; ++r)
    C[(size_t)(r0 + r) * N + col] = acc[r] + bv;
}

// ---------------- fragment index helpers (device) ----------------
// A-fragment element (kt, mt, lane, i) -> row = mt*16 + lane%16,
//   k = kt*32 + (lane/16)*8 + (i&7) + (i>=8 ? 16 : 0)
// B-fragment element (kt, nt, lane, i) -> col = nt*16 + lane%16,
//   k = kt*32 + (lane/16)*8 + (i&7) + (i>=8 ? 16 : 0)

// Pack f32 row-major A (M x K) into A-fragment bf16 layout.
__global__ void apack_kernel(const float* __restrict__ A, bf16_t* __restrict__ dst,
                             int M, int K) {
  int idx = blockIdx.x * blockDim.x + threadIdx.x;
  if (idx >= M * K) return;
  int i = idx & 15, r = idx >> 4;
  int lane = r & 31; r >>= 5;
  int Mt = M >> 4;
  int mt = r % Mt, kt = r / Mt;
  int row = mt * 16 + (lane & 15);
  int k = kt * 32 + ((lane >> 4) << 3) + (i & 7) + ((i >> 3) << 4);
  dst[idx] = (bf16_t)A[(size_t)row * K + k];
}

// Pack concat([s0|s1|s2], each 32 x nX f32) into A-fragment bf16 layout (M=32).
__global__ void apack3_kernel(const float* __restrict__ s0, int n0,
                              const float* __restrict__ s1, int n1,
                              const float* __restrict__ s2, int n2,
                              bf16_t* __restrict__ dst) {
  int K = n0 + n1 + n2;
  int idx = blockIdx.x * blockDim.x + threadIdx.x;
  if (idx >= B_ * K) return;
  int i = idx & 15, r = idx >> 4;
  int lane = r & 31; r >>= 5;
  int mt = r & 1, kt = r >> 1;                 // M = 32 -> 2 row tiles
  int row = mt * 16 + (lane & 15);
  int k = kt * 32 + ((lane >> 4) << 3) + (i & 7) + ((i >> 3) << 4);
  float v;
  if (k < n0)            v = s0[row * n0 + k];
  else if (k < n0 + n1)  v = s1[row * n1 + (k - n0)];
  else                   v = s2[row * n2 + (k - n0 - n1)];
  dst[idx] = (bf16_t)v;
}

// Pack weight stack [W0 (K0 x N); W1 ((K-K0) x N)] (f32 row-major) into
// B-fragment bf16 layout.  For a single matrix pass K0 = K, W1 = W0.
__global__ void wpack_kernel(const float* __restrict__ W0, int K0,
                             const float* __restrict__ W1,
                             bf16_t* __restrict__ dst, int K, int N) {
  int idx = blockIdx.x * blockDim.x + threadIdx.x;
  if (idx >= K * N) return;
  int i = idx & 15, r = idx >> 4;
  int lane = r & 31; r >>= 5;
  int Nt = N >> 4;
  int nt = r % Nt, kt = r / Nt;
  int col = nt * 16 + (lane & 15);
  int k = kt * 32 + ((lane >> 4) << 3) + (i & 7) + ((i >> 3) << 4);
  float v = (k < K0) ? W0[(size_t)k * N + col] : W1[(size_t)(k - K0) * N + col];
  dst[idx] = (bf16_t)v;
}

// --------------------------- small helper kernels ---------------------------
__global__ void addvec_kernel(const float* __restrict__ a, const float* __restrict__ b,
                              float* __restrict__ o, int n) {
  int i = blockIdx.x * blockDim.x + threadIdx.x;
  if (i < n) o[i] = a[i] + b[i];
}

__global__ void zerof_kernel(float* __restrict__ p, int n) {
  int i = blockIdx.x * blockDim.x + threadIdx.x;
  if (i < n) p[i] = 0.f;
}

__device__ inline float sigm(float x) { return 1.f / (1.f + expf(-x)); }

// g: (32 x 4H) gate pre-activations; updates h, c in place.  H = 1024.
__global__ void lstm_update_kernel(const float* __restrict__ g,
                                   float* __restrict__ h, float* __restrict__ c, int H) {
  int idx = blockIdx.x * blockDim.x + threadIdx.x;
  if (idx >= B_ * H) return;
  int b = idx / H, n = idx % H;
  const float* gr = g + (size_t)b * 4 * H;
  float i  = gr[n];
  float f  = gr[n + H];
  float gg = gr[n + 2 * H];
  float o  = gr[n + 3 * H];
  float c2 = sigm(f) * c[idx] + sigm(i) * tanhf(gg);
  c[idx] = c2;
  h[idx] = sigm(o) * tanhf(c2);
}

// Prenet: x[t,b,:] = relu(relu(dec_in[t,b,:] @ W1 + b1) @ W2 + b2)
// dec_in[0]=0, dec_in[t]=decoder_inputs[:,:,t-1].  decoder_inputs: (B,256,200)
__global__ __launch_bounds__(128) void prenet_kernel(
    const float* __restrict__ dec, const float* __restrict__ W1,
    const float* __restrict__ b1, const float* __restrict__ W2,
    const float* __restrict__ b2, float* __restrict__ x) {
  int t = blockIdx.x / B_, b = blockIdx.x % B_, j = threadIdx.x;
  __shared__ float h1[PRE];
  float s = b1[j];
  if (t > 0) {
    const float* di = dec + (size_t)b * HID * TOUT + (t - 1);
    for (int h = 0; h < HID; ++h) s += di[(size_t)h * TOUT] * W1[h * PRE + j];
  }
  h1[j] = fmaxf(s, 0.f);
  __syncthreads();
  float s2 = b2[j];
  for (int k = 0; k < PRE; ++k) s2 += h1[k] * W2[k * PRE + j];
  x[((size_t)t * B_ + b) * PRE + j] = fmaxf(s2, 0.f);
}

// Location-sensitive attention: one block per batch row, 256 threads (one per
// encoder timestep).  Updates aw, awc, ctx; writes this step's alignment row.
__global__ __launch_bounds__(256) void attention_kernel(
    const float* __restrict__ ah, const float* __restrict__ proc_mem,
    const float* __restrict__ enc, const int* __restrict__ memlen,
    const float* __restrict__ Wq, const float* __restrict__ Wv,
    const float* __restrict__ Wconv, const float* __restrict__ Wloc,
    float* __restrict__ aw, float* __restrict__ awc, float* __restrict__ ctx,
    float* __restrict__ aligns, int tstep) {
  int b = blockIdx.x, t = threadIdx.x;
  __shared__ float q[ATT];
  __shared__ float aw_s[TIN], awc_s[TIN], sm[TIN];

  aw_s[t]  = aw[b * TIN + t];
  awc_s[t] = awc[b * TIN + t];
  if (t < ATT) {                              // q = ah @ Wq  (row b)
    float s = 0.f;
    const float* ap = ah + (size_t)b * ARNN;
    for (int k = 0; k < ARNN; ++k) s += ap[k] * Wq[k * ATT + t];
    q[t] = s;
  }
  __syncthreads();

  // location conv: 32 filters over [aw; awc], kernel 31, pad 15
  float convout[NF];
#pragma unroll 4
  for (int f = 0; f < NF; ++f) {
    float s = 0.f;
    const float* wf = Wconv + f * 2 * KS;
    for (int k = 0; k < KS; ++k) {
      int idx = t + k - (KS - 1) / 2;
      if (idx >= 0 && idx < TIN)
        s += aw_s[idx] * wf[k] + awc_s[idx] * wf[KS + k];
    }
    convout[f] = s;
  }
  // energy_t = sum_a tanh(q[a] + proc_mem[b,t,a] + loc[a]) * Wv[a]
  float e = 0.f;
  const float* pm = proc_mem + ((size_t)b * TIN + t) * ATT;
  for (int a = 0; a < ATT; ++a) {
    float l = 0.f;
#pragma unroll 8
    for (int f = 0; f < NF; ++f) l += convout[f] * Wloc[f * ATT + a];
    e += tanhf(q[a] + pm[a] + l) * Wv[a];
  }
  if (t >= memlen[b]) e = -__builtin_inff();

  // softmax over 256 threads
  sm[t] = e; __syncthreads();
  for (int s = 128; s > 0; s >>= 1) { if (t < s) sm[t] = fmaxf(sm[t], sm[t + s]); __syncthreads(); }
  float mx = sm[0]; __syncthreads();
  float ex = expf(e - mx);
  sm[t] = ex; __syncthreads();
  for (int s = 128; s > 0; s >>= 1) { if (t < s) sm[t] += sm[t + s]; __syncthreads(); }
  float p = ex / sm[0];

  aw_s[t] = p;
  aw[b * TIN + t]  = p;
  awc[b * TIN + t] = awc_s[t] + p;
  aligns[((size_t)b * TOUT + tstep) * TIN + t] = p;
  __syncthreads();

  // ctx[b,e] = sum_t aw[t] * enc[b,t,e]   (thread t computes channel e = t)
  float c = 0.f;
  const float* eb = enc + (size_t)b * TIN * EMB + t;
  for (int tin = 0; tin < TIN; ++tin) c += aw_s[tin] * eb[(size_t)tin * EMB];
  ctx[b * EMB + t] = c;
}

// Scatter projection (32x256, f32) into outs[b, h, t]
__global__ void scatter_out_kernel(const float* __restrict__ proj,
                                   float* __restrict__ outs, int t) {
  int idx = blockIdx.x * blockDim.x + threadIdx.x;   // 8192
  int b = idx >> 8, n = idx & 255;
  outs[((size_t)b * HID + n) * TOUT + t] = proj[idx];
}

// stops[b, t] = [dh | ctx] @ Wgate + bgate
__global__ void gate_kernel(const float* __restrict__ dh, const float* __restrict__ ctx,
                            const float* __restrict__ Wg, const float* __restrict__ bg,
                            float* __restrict__ stops, int t) {
  int b = threadIdx.x;
  if (b >= B_) return;
  float s = bg[0];
  const float* dr = dh + (size_t)b * DRNN;
  for (int k = 0; k < DRNN; ++k) s += dr[k] * Wg[k];
  const float* cr = ctx + (size_t)b * EMB;
  for (int e = 0; e < EMB; ++e) s += cr[e] * Wg[DRNN + e];
  stops[b * TOUT + t] = s;
}

// ---------------------------------------------------------------------------
extern "C" void kernel_launch(void* const* d_in, const int* in_sizes, int n_in,
                              void* d_out, int out_size, void* d_ws, size_t ws_size,
                              hipStream_t stream) {
  (void)in_sizes; (void)n_in; (void)out_size; (void)ws_size;
  const float* enc   = (const float*)d_in[0];
  const float* dec   = (const float*)d_in[1];
  const int*   mlen  = (const int*)d_in[2];
  const float* Wmem  = (const float*)d_in[3];
  const float* Wp1   = (const float*)d_in[4];
  const float* bp1   = (const float*)d_in[5];
  const float* Wp2   = (const float*)d_in[6];
  const float* bp2   = (const float*)d_in[7];
  const float* Wih_a = (const float*)d_in[8];
  const float* Whh_a = (const float*)d_in[9];
  const float* bih_a = (const float*)d_in[10];
  const float* bhh_a = (const float*)d_in[11];
  const float* Wq    = (const float*)d_in[12];
  const float* Wv    = (const float*)d_in[13];
  const float* Wconv = (const float*)d_in[14];
  const float* Wloc  = (const float*)d_in[15];
  const float* Wih_d = (const float*)d_in[16];
  const float* Whh_d = (const float*)d_in[17];
  const float* bih_d = (const float*)d_in[18];
  const float* bhh_d = (const float*)d_in[19];
  const float* Wproj = (const float*)d_in[20];
  const float* bprj  = (const float*)d_in[21];
  const float* Wgate = (const float*)d_in[22];
  const float* bgate = (const float*)d_in[23];

  // ---- workspace layout (bump allocator, 256B aligned) ----
  char* base = (char*)d_ws;
  size_t off = 0;
  auto alloc = [&](size_t bytes) -> char* {
    char* p = base + off;
    off = (off + bytes + 255) & ~(size_t)255;
    return p;
  };
  const int KA = PRE + EMB + ARNN;   // 1408
  const int KD = ARNN + EMB + DRNN;  // 2304
  const int KP = DRNN + EMB;         // 1280

  bf16_t* wcat_a   = (bf16_t*)alloc((size_t)KA * 4 * ARNN * sizeof(bf16_t));
  bf16_t* wcat_d   = (bf16_t*)alloc((size_t)KD * 4 * DRNN * sizeof(bf16_t));
  bf16_t* wproj_b  = (bf16_t*)alloc((size_t)KP * HID * sizeof(bf16_t));
  bf16_t* wmem_b   = (bf16_t*)alloc((size_t)EMB * ATT * sizeof(bf16_t));
  bf16_t* enc_frag = (bf16_t*)alloc((size_t)B_ * TIN * EMB * sizeof(bf16_t));
  float*  bias_a   = (float*)alloc(4 * ARNN * sizeof(float));
  float*  bias_d   = (float*)alloc(4 * DRNN * sizeof(float));
  float*  zbias    = (float*)alloc(ATT * sizeof(float));
  float*  xpre     = (float*)alloc((size_t)TOUT * B_ * PRE * sizeof(float));
  float*  procmem  = (float*)alloc((size_t)B_ * TIN * ATT * sizeof(float));
  float*  states   = (float*)alloc((size_t)(4 * B_ * ARNN + 3 * B_ * TIN) * sizeof(float));
  float*  ah  = states;
  float*  ac  = ah + B_ * ARNN;
  float*  dh  = ac + B_ * ARNN;
  float*  dc  = dh + B_ * DRNN;
  float*  aw  = dc + B_ * DRNN;
  float*  awc = aw + B_ * TIN;
  float*  ctx = awc + B_ * TIN;
  const int STATE_N = 4 * B_ * ARNN + 3 * B_ * TIN;
  bf16_t* acat_a  = (bf16_t*)alloc((size_t)B_ * KA * sizeof(bf16_t));
  bf16_t* acat_d  = (bf16_t*)alloc((size_t)B_ * KD * sizeof(bf16_t));
  bf16_t* hcbuf   = (bf16_t*)alloc((size_t)B_ * KP * sizeof(bf16_t));
  float*  gbuf    = (float*)alloc((size_t)B_ * 4 * ARNN * sizeof(float));
  float*  projout = (float*)alloc((size_t)B_ * HID * sizeof(float));

  float* outs   = (float*)d_out;                        // (B, HID, TOUT)
  float* stops  = outs + (size_t)B_ * HID * TOUT;       // (B, TOUT)
  float* aligns = stops + (size_t)B_ * TOUT;            // (B, TOUT, TIN)

  auto cdiv = [](int a, int b) { return (a + b - 1) / b; };

  // ---- prologue ----
  zerof_kernel<<<cdiv(STATE_N, 256), 256, 0, stream>>>(states, STATE_N);
  zerof_kernel<<<1, ATT, 0, stream>>>(zbias, ATT);
  // weight packs into B-fragment layout
  wpack_kernel<<<cdiv(KA * 4 * ARNN, 256), 256, 0, stream>>>(
      Wih_a, (PRE + EMB), Whh_a, wcat_a, KA, 4 * ARNN);
  wpack_kernel<<<cdiv(KD * 4 * DRNN, 256), 256, 0, stream>>>(
      Wih_d, (ARNN + EMB), Whh_d, wcat_d, KD, 4 * DRNN);
  wpack_kernel<<<cdiv(KP * HID, 256), 256, 0, stream>>>(
      Wproj, KP, Wproj, wproj_b, KP, HID);
  wpack_kernel<<<cdiv(EMB * ATT, 256), 256, 0, stream>>>(
      Wmem, EMB, Wmem, wmem_b, EMB, ATT);
  // encoder_out (8192 x 256) into A-fragment layout
  apack_kernel<<<cdiv(B_ * TIN * EMB, 256), 256, 0, stream>>>(
      enc, enc_frag, B_ * TIN, EMB);
  addvec_kernel<<<cdiv(4 * ARNN, 256), 256, 0, stream>>>(bih_a, bhh_a, bias_a, 4 * ARNN);
  addvec_kernel<<<cdiv(4 * DRNN, 256), 256, 0, stream>>>(bih_d, bhh_d, bias_d, 4 * DRNN);
  prenet_kernel<<<TOUT * B_, PRE, 0, stream>>>(dec, Wp1, bp1, Wp2, bp2, xpre);
  // proc_mem = encoder_out (8192 x 256) @ W_mem (256 x 128)
  wmma_gemm<<<dim3(cdiv(ATT / 16, 4), (B_ * TIN) / 16), 128, 0, stream>>>(
      enc_frag, wmem_b, zbias, procmem, B_ * TIN, ATT, EMB);

  // ---- 200 sequential decode steps ----
  for (int t = 0; t < TOUT; ++t) {
    // attention LSTM: A = [x_t | ctx | ah], W = [Wih_a ; Whh_a]
    apack3_kernel<<<cdiv(B_ * KA, 256), 256, 0, stream>>>(
        xpre + (size_t)t * B_ * PRE, PRE, ctx, EMB, ah, ARNN, acat_a);
    wmma_gemm<<<dim3((4 * ARNN / 16) / 4, B_ / 16), 128, 0, stream>>>(
        acat_a, wcat_a, bias_a, gbuf, B_, 4 * ARNN, KA);
    lstm_update_kernel<<<cdiv(B_ * ARNN, 256), 256, 0, stream>>>(gbuf, ah, ac, ARNN);

    attention_kernel<<<B_, TIN, 0, stream>>>(
        ah, procmem, enc, mlen, Wq, Wv, Wconv, Wloc, aw, awc, ctx, aligns, t);

    // decoder LSTM: A = [ah | ctx | dh], W = [Wih_d ; Whh_d]
    apack3_kernel<<<cdiv(B_ * KD, 256), 256, 0, stream>>>(
        ah, ARNN, ctx, EMB, dh, DRNN, acat_d);
    wmma_gemm<<<dim3((4 * DRNN / 16) / 4, B_ / 16), 128, 0, stream>>>(
        acat_d, wcat_d, bias_d, gbuf, B_, 4 * DRNN, KD);
    lstm_update_kernel<<<cdiv(B_ * DRNN, 256), 256, 0, stream>>>(gbuf, dh, dc, DRNN);

    // projection + gate
    apack3_kernel<<<cdiv(B_ * KP, 256), 256, 0, stream>>>(
        dh, DRNN, ctx, EMB, dh, 0, hcbuf);
    wmma_gemm<<<dim3((HID / 16) / 4, B_ / 16), 128, 0, stream>>>(
        hcbuf, wproj_b, bprj, projout, B_, HID, KP);
    scatter_out_kernel<<<(B_ * HID) / 256, 256, 0, stream>>>(projout, outs, t);
    gate_kernel<<<1, 32, 0, stream>>>(dh, ctx, Wgate, bgate, stops, t);
  }
}
